// Head_41308995453135
// MI455X (gfx1250) — matrix operand order repaired and checked by reading, not compiled, not run on hardware
//
#include <hip/hip_runtime.h>
#include <hip/hip_bf16.h>

// ---------------------------------------------------------------------------
// Single-head causal attention, B=4, T=4096, C=1024, H=64 (fp32 in/out).
// bf16 WMMA (v_wmma_f32_16x16x32_bf16) + flash-attention online softmax.
// QKV GEMM: copy-free rotating software pipeline (K-loop unrolled by 2);
// __launch_bounds__(.,1) lifts the default 128-VGPR cap so accumulators and
// in-flight fragments live in registers (no scratch spills).
// ---------------------------------------------------------------------------

typedef __attribute__((ext_vector_type(16))) __bf16 v16bf;
typedef __attribute__((ext_vector_type(8)))  __bf16 v8bf;
typedef __attribute__((ext_vector_type(8)))  float  v8f;
typedef __attribute__((ext_vector_type(4)))  float  v4f;

#define T_SEQ 4096
#define C_DIM 1024
#define H_DIM 64

static __device__ __forceinline__ v8f wmma_bf16(v16bf a, v16bf b, v8f c) {
    return __builtin_amdgcn_wmma_f32_16x16x32_bf16(
        /*neg_a=*/false, a, /*neg_b=*/false, b,
        /*c_mod=*/(short)0, c, /*reuse_a=*/false, /*reuse_b=*/false);
}

static __device__ __forceinline__ v8bf ld8bf(const __bf16* p) {
    return *reinterpret_cast<const v8bf*>(p);
}

static __device__ __forceinline__ v16bf pack16(v8bf lo, v8bf hi) {
    return __builtin_shufflevector(lo, hi, 0,1,2,3,4,5,6,7,8,9,10,11,12,13,14,15);
}

// Convert 8 consecutive fp32 (two float4 loads) -> v8bf
static __device__ __forceinline__ v8bf cvt8(const float* p) {
    v4f a = *reinterpret_cast<const v4f*>(p);
    v4f b = *reinterpret_cast<const v4f*>(p + 4);
    v8bf r;
#pragma unroll
    for (int i = 0; i < 4; i++) { r[i] = (__bf16)a[i]; r[4 + i] = (__bf16)b[i]; }
    return r;
}

// A fragment (16x32 bf16) of x: lanes 0-15 hold K {0..15},
// lanes 16-31 hold K {16..31}.
static __device__ __forceinline__ v16bf loadA(const float* xr, int k0, int koff) {
    return pack16(cvt8(xr + k0 + koff), cvt8(xr + k0 + 8 + koff));
}

// One B fragment (i = m*4 + nt) from K-major bf16 weights.
static __device__ __forceinline__ v16bf loadBfrag(const __bf16* Wb, int k0, int i) {
    const __bf16* wb = Wb + ((size_t)i << 14) + k0;   // i*16*1024 elements
    return pack16(ld8bf(wb), ld8bf(wb + 8));
}

// ---------------------------------------------------------------------------
// Kernel 1: Wt[m][n][c] = W_m[c][n] as bf16 (K-major weights => contiguous
// B-fragment loads in the projection GEMM).
// ---------------------------------------------------------------------------
__global__ void wt_kernel(const float* __restrict__ Wq,
                          const float* __restrict__ Wk,
                          const float* __restrict__ Wv,
                          __bf16* __restrict__ Wt) {
    int idx = blockIdx.x * blockDim.x + threadIdx.x;       // [0, 3*64*1024)
    if (idx >= 3 * H_DIM * C_DIM) return;
    int m = idx >> 16;                // which matrix
    int rem = idx & 0xFFFF;
    int n = rem >> 10;                // head col
    int c = rem & 1023;               // embed row
    const float* W = (m == 0) ? Wq : (m == 1) ? Wk : Wv;
    Wt[idx] = (__bf16)W[c * H_DIM + n];
}

// ---------------------------------------------------------------------------
// Kernel 2: fused QKV projection, copy-free rotating pipeline.
// Block = 128 (4 waves), each wave owns a 16-row tile of x; 12 WMMAs
// (q,k,v x 4 n-tiles) per 32-wide K-step; K-loop unrolled by 2 so the two
// A registers and the b[] slots alternate roles with no move chains.
// ---------------------------------------------------------------------------
__global__ void __launch_bounds__(128, 1)
qkv_kernel(const float* __restrict__ x,
           const __bf16* __restrict__ Wt,
           const float* __restrict__ bq,
           const float* __restrict__ bk,
           const float* __restrict__ bv,
           __bf16* __restrict__ qo,
           __bf16* __restrict__ ko,
           __bf16* __restrict__ vto) {
    const int lane = threadIdx.x & 31;
    const int wave = threadIdx.x >> 5;
    const int ln   = lane & 15;
    const int half = lane >> 4;

    const int row0 = blockIdx.x * 64 + wave * 16;   // global row tile
    const int arow = row0 + ln;                     // this lane's A row
    const int koff = half ? 16 : 0;                 // A-layout K offset
    const int ksel = half * 16;                     // B-layout K offset

    const float*  xr = x + (size_t)arow * C_DIM;
    const __bf16* Wb = Wt + ((size_t)ln << 10) + ksel;  // per-lane weight base

    v8f acc[12];
#pragma unroll
    for (int i = 0; i < 12; i++) acc[i] = (v8f){};

    // Prologue: stage K-step 0.
    v16bf a0 = loadA(xr, 0, koff);
    v16bf a1;
    v16bf b[12];
#pragma unroll
    for (int i = 0; i < 12; i++) b[i] = loadBfrag(Wb, 0, i);

#pragma unroll 1
    for (int k0 = 0; k0 < C_DIM; k0 += 64) {
        // Substep 0: compute with (a0, b[]), refill slots for k0+32.
        a1 = loadA(xr, k0 + 32, koff);
#pragma unroll
        for (int i = 0; i < 12; i++) {
            acc[i] = wmma_bf16(a0, b[i], acc[i]);
            b[i] = loadBfrag(Wb, k0 + 32, i);
        }
        // Substep 1: compute with (a1, b[]), refill slots for k0+64
        // (clamped on the final iteration; harmless redundant load).
        const int kn = (k0 + 64 < C_DIM) ? (k0 + 64) : 0;
        a0 = loadA(xr, kn, koff);
#pragma unroll
        for (int i = 0; i < 12; i++) {
            acc[i] = wmma_bf16(a1, b[i], acc[i]);
            b[i] = loadBfrag(Wb, kn, i);
        }
    }

    // Epilogue: bias add + store. C/D layout: VGPR j -> row j (lanes 0-15)
    // or row 8+j (lanes 16-31); N = lane&15 within the n-tile.
#pragma unroll
    for (int nt = 0; nt < 4; nt++) {
        const int n = nt * 16 + ln;
        const float bqv = bq[n], bkv = bk[n], bvv = bv[n];
#pragma unroll
        for (int j = 0; j < 8; j++) {
            const int row = row0 + j + half * 8;
            qo[(size_t)row * H_DIM + n] = (__bf16)(acc[0 + nt][j] + bqv);
            ko[(size_t)row * H_DIM + n] = (__bf16)(acc[4 + nt][j] + bkv);
            const int batch = row >> 12;          // T = 4096
            const int t     = row & (T_SEQ - 1);
            vto[((size_t)batch * H_DIM + n) * T_SEQ + t] =
                (__bf16)(acc[8 + nt][j] + bvv);
        }
    }
}

// ---------------------------------------------------------------------------
// Kernel 3: flash attention. Block = 256 (8 waves), each wave owns one
// 16-query tile, loops causally over 32 key columns per step:
//   load K and V fragments up front (V overlaps the softmax chain)
//   4 WMMAs  S = Q K^T   (2 seq n-tiles x 2 k-steps over H=64)
//   online softmax (shfl_xor reductions inside 16-lane halves)
//   P -> LDS -> A-fragment relayout (same-wave, s_wait_dscnt)
//   4 WMMAs  O += P V    (4 head n-tiles, K=32 seq)
// ---------------------------------------------------------------------------
__global__ void __launch_bounds__(256, 1)
attn_kernel(const __bf16* __restrict__ q,
            const __bf16* __restrict__ k,
            const __bf16* __restrict__ vt,
            float* __restrict__ out) {
    __shared__ __bf16 Pbuf[8][16 * 32];

    const int lane = threadIdx.x & 31;
    const int wave = threadIdx.x >> 5;
    const int ln   = lane & 15;
    const int half = lane >> 4;
    const int koff = half ? 16 : 0;

    const int b    = blockIdx.y;
    const int row0 = (blockIdx.x * 8 + wave) * 16;   // query tile (in batch)

    const __bf16* qb = q  + (size_t)b * T_SEQ * H_DIM;
    const __bf16* kb = k  + (size_t)b * T_SEQ * H_DIM;
    const __bf16* vb = vt + (size_t)b * H_DIM * T_SEQ;

    // Q A-fragments for the two 32-wide k-steps over H=64.
    v16bf aq[2];
    {
        const __bf16* qr = qb + (size_t)(row0 + ln) * H_DIM;
#pragma unroll
        for (int h = 0; h < 2; h++)
            aq[h] = pack16(ld8bf(qr + h * 32 + koff),
                           ld8bf(qr + h * 32 + 8 + koff));
    }

    float m_i[8], l_i[8];
    v8f o[4];
#pragma unroll
    for (int j = 0; j < 8; j++) { m_i[j] = -1e30f; l_i[j] = 0.0f; }
#pragma unroll
    for (int nt = 0; nt < 4; nt++) o[nt] = (v8f){};

    const float scale = 0.03125f;   // C^-0.5, C = 1024
    const int s_end = row0 + 16;    // causal bound for this tile

    for (int s0 = 0; s0 < s_end; s0 += 32) {
        // ---- stage all K and V fragments for this step up front ----
        v16bf bkf[2][2];
#pragma unroll
        for (int nt = 0; nt < 2; nt++) {
            const __bf16* kr = kb + (size_t)(s0 + nt * 16 + ln) * H_DIM + half * 16;
            bkf[nt][0] = pack16(ld8bf(kr),      ld8bf(kr + 8));
            bkf[nt][1] = pack16(ld8bf(kr + 32), ld8bf(kr + 40));
        }
        v16bf bvf[4];
#pragma unroll
        for (int nt = 0; nt < 4; nt++) {
            const __bf16* vr = vb + (size_t)(nt * 16 + ln) * T_SEQ + s0 + half * 16;
            bvf[nt] = pack16(ld8bf(vr), ld8bf(vr + 8));   // overlaps softmax
        }

        // ---- S = Q K^T for 32 key columns (two 16-col fragments) ----
        v8f s_f[2];
#pragma unroll
        for (int nt = 0; nt < 2; nt++) {
            v8f sa = (v8f){};
            sa = wmma_bf16(aq[0], bkf[nt][0], sa);
            sa = wmma_bf16(aq[1], bkf[nt][1], sa);
            s_f[nt] = sa;
        }

        // ---- scale + causal mask + online softmax ----
        float mnew[8];
#pragma unroll
        for (int j = 0; j < 8; j++) {
            const int row = row0 + j + half * 8;
#pragma unroll
            for (int nt = 0; nt < 2; nt++) {
                const int col = s0 + nt * 16 + ln;
                float v = s_f[nt][j] * scale;
                if (col > row) v = -1e30f;
                s_f[nt][j] = v;
            }
            float rm = fmaxf(s_f[0][j], s_f[1][j]);
            rm = fmaxf(rm, __shfl_xor(rm, 1));
            rm = fmaxf(rm, __shfl_xor(rm, 2));
            rm = fmaxf(rm, __shfl_xor(rm, 4));
            rm = fmaxf(rm, __shfl_xor(rm, 8));
            mnew[j] = fmaxf(m_i[j], rm);
        }

#pragma unroll
        for (int j = 0; j < 8; j++) {
            const float corr = __expf(m_i[j] - mnew[j]);
            const float p0 = __expf(s_f[0][j] - mnew[j]);
            const float p1 = __expf(s_f[1][j] - mnew[j]);
            s_f[0][j] = p0; s_f[1][j] = p1;
            float rs = p0 + p1;
            rs += __shfl_xor(rs, 1);
            rs += __shfl_xor(rs, 2);
            rs += __shfl_xor(rs, 4);
            rs += __shfl_xor(rs, 8);
            l_i[j] = l_i[j] * corr + rs;
            m_i[j] = mnew[j];
#pragma unroll
            for (int nt = 0; nt < 4; nt++) o[nt][j] *= corr;
        }

        // ---- P relayout: C-layout regs -> LDS -> A-layout fragment ----
        __bf16* Pw = &Pbuf[wave][0];
#pragma unroll
        for (int j = 0; j < 8; j++) {
            const int r = j + half * 8;
            Pw[r * 32 + ln]      = (__bf16)s_f[0][j];
            Pw[r * 32 + 16 + ln] = (__bf16)s_f[1][j];
        }
        asm volatile("s_wait_dscnt 0" ::: "memory");   // same-wave RAW on LDS
        v16bf pa = pack16(*reinterpret_cast<const v8bf*>(Pw + ln * 32 + koff),
                          *reinterpret_cast<const v8bf*>(Pw + ln * 32 + 8 + koff));

        // ---- O += P V  (V fragments already resident) ----
#pragma unroll
        for (int nt = 0; nt < 4; nt++)
            o[nt] = wmma_bf16(pa, bvf[nt], o[nt]);
    }

    // ---- normalize and store fp32 output [B,T,64] ----
    float* ob = out + (size_t)b * T_SEQ * H_DIM;
#pragma unroll
    for (int nt = 0; nt < 4; nt++) {
        const int n = nt * 16 + ln;
#pragma unroll
        for (int j = 0; j < 8; j++) {
            const int row = row0 + j + half * 8;
            ob[(size_t)row * H_DIM + n] = o[nt][j] / l_i[j];
        }
    }
}

// ---------------------------------------------------------------------------
extern "C" void kernel_launch(void* const* d_in, const int* in_sizes, int n_in,
                              void* d_out, int out_size, void* d_ws, size_t ws_size,
                              hipStream_t stream) {
    const float* x  = (const float*)d_in[0];
    const float* Wq = (const float*)d_in[1];
    const float* bq = (const float*)d_in[2];
    const float* Wk = (const float*)d_in[3];
    const float* bk = (const float*)d_in[4];
    const float* Wv = (const float*)d_in[5];
    const float* bv = (const float*)d_in[6];
    float* out = (float*)d_out;

    const int BT = in_sizes[0] / C_DIM;       // B*T = 16384
    const int B  = BT / T_SEQ;                // 4

    // Workspace layout
    char* ws = (char*)d_ws;
    __bf16* Wt   = (__bf16*)ws;                                  // 3*64*1024
    __bf16* qbuf = (__bf16*)(ws + (size_t)3 * H_DIM * C_DIM * 2);
    __bf16* kbuf = qbuf + (size_t)BT * H_DIM;
    __bf16* vtbf = kbuf + (size_t)BT * H_DIM;

    // 1) weight transpose + bf16 convert
    wt_kernel<<<(3 * H_DIM * C_DIM + 255) / 256, 256, 0, stream>>>(Wq, Wk, Wv, Wt);

    // 2) fused QKV projection (wave per 16-row tile, 4 waves per block)
    qkv_kernel<<<BT / 64, 128, 0, stream>>>(x, Wt, bq, bk, bv, qbuf, kbuf, vtbf);

    // 3) flash attention (wave per 16-query tile, 8 waves per block)
    dim3 grid(T_SEQ / 128, B);
    attn_kernel<<<grid, 256, 0, stream>>>(qbuf, kbuf, vtbf, out);
}
